// TransformerBlock_53145925320720
// MI455X (gfx1250) — matrix operand order repaired and checked by reading, not compiled, not run on hardware
//
#include <hip/hip_runtime.h>
#include <math.h>

// ---------------------------------------------------------------------------
// Transformer block for MI455X (gfx1250): all GEMMs via v_wmma_f32_16x16x32_bf16.
// Weights pre-transposed once (cast+transpose) so every GEMM stages B with
// contiguous b128 LDS writes; staging uses GLOBAL_LOAD_ASYNC_TO_LDS_B128 via
// inline asm (ASYNCcnt path, bypasses clang builtin address-space quirks).
// ---------------------------------------------------------------------------

#define BDIM 256
#define USE_ASYNC_LDS 1

typedef __bf16 bf16_t;
typedef __attribute__((ext_vector_type(16))) __bf16         v16bf;
typedef __attribute__((ext_vector_type(8)))  float          v8f;
typedef __attribute__((ext_vector_type(4)))  float          f32x4;
typedef __attribute__((ext_vector_type(4)))  unsigned int   u32x4;
typedef __attribute__((ext_vector_type(4)))  unsigned short ushort4_t;

#if USE_ASYNC_LDS
// GV-mode async copy: 16 bytes global -> LDS, tracked by ASYNCcnt.
// vdst = VGPR with LDS byte address (low 32 bits of generic AS3 pointer),
// vaddr = 64-bit global address VGPR pair, saddr = off.
__device__ __forceinline__ void async_cp16(const unsigned short* g, unsigned short* l) {
  asm volatile("global_load_async_to_lds_b128 %0, %1, off"
               :: "v"((unsigned)(unsigned long long)l), "v"(g)
               : "memory");
}
__device__ __forceinline__ void wait_async0() {
  asm volatile("s_wait_asynccnt 0x0" ::: "memory");
}
#endif

static constexpr int Bb = 2, Ss = 2048, Hh = 1024, NHh = 16, DHh = 64, FFf = 4096;

__device__ __forceinline__ unsigned short f2bf(float f) {
  union { bf16_t b; unsigned short u; } cv; cv.b = (bf16_t)f; return cv.u;
}

// --------------------------- fp32 -> bf16 cast ------------------------------
__global__ __launch_bounds__(BDIM) void cast_f32_bf16(const float* __restrict__ in,
                                                      unsigned short* __restrict__ out,
                                                      int n4) {
  int i = blockIdx.x * BDIM + threadIdx.x;
  int stride = gridDim.x * BDIM;
  for (; i < n4; i += stride) {
    f32x4 v = ((const f32x4*)in)[i];
    ushort4_t o = { f2bf(v.x), f2bf(v.y), f2bf(v.z), f2bf(v.w) };
    ((ushort4_t*)out)[i] = o;
  }
}

// ---------------- fp32 [R][C] -> bf16 [C][R] (cast + transpose) -------------
__global__ __launch_bounds__(BDIM) void cast_transpose(const float* __restrict__ in,
                                                       unsigned short* __restrict__ out,
                                                       int R, int C) {
  __shared__ unsigned short tile[32][33];
  const int bx = blockIdx.x * 32;       // col base in source
  const int by = blockIdx.y * 32;       // row base in source
  const int tx = threadIdx.x & 31;
  const int ty = threadIdx.x >> 5;      // 0..7
#pragma unroll
  for (int i = 0; i < 32; i += 8)
    tile[ty + i][tx] = f2bf(in[(long long)(by + ty + i) * C + bx + tx]);
  __syncthreads();
#pragma unroll
  for (int i = 0; i < 32; i += 8)
    out[(long long)(bx + ty + i) * R + by + tx] = tile[tx][ty + i];
}

// --------------------------- WMMA GEMM --------------------------------------
// C[z][M,N] = scale * A[z][M,K] * B^T  (+bias) (+exact GELU)
// B is ALWAYS given as [N][K] rows (pre-transposed weights / K-matrix / V^T).
// A_F32=1 : A is fp32, converted to bf16 while staging into LDS
// OUT     : 0 = fp32, 1 = bf16, 2 = bf16 stored transposed (C^T[N][M])
// EPI     : 0 = none, 1 = +bias, 2 = +bias then exact GELU
template <int BN, int EPI, int OUT, int A_F32>
__global__ __launch_bounds__(BDIM) void gemm_wmma(
    const void* __restrict__ Ap, const unsigned short* __restrict__ Bp,
    const float* __restrict__ bias, void* __restrict__ Cp,
    int K, int lda, int ldb, int ldc,
    long long a_bstr, long long a_hstr,
    long long b_bstr, long long b_hstr,
    long long c_bstr, long long c_hstr,
    int nh, float scale) {
  constexpr int BM = 128, BK = 64;
  constexpr int LSTR = BK + 8;                 // u16 units; 144B rows, 16B aligned
  constexpr int WGN = (BN == 128) ? 2 : 1;
  constexpr int WGM = 8 / WGN;
  constexpr int WTM = BM / WGM;                // 32 or 16 rows per wave
  constexpr int ITM = WTM / 16;
  constexpr int ITN = 4;                       // 64 cols per wave
  constexpr int ACH = (BM * BK) / (BDIM * 8);  // 4
  constexpr int BCH = (BN * BK) / (BDIM * 8);  // 4 or 2

  __shared__ unsigned short sA[BM * LSTR];
  __shared__ unsigned short sB[BN * LSTR];

  const int tid  = threadIdx.x;
  const int lane = tid & 31;
  const int wv   = tid >> 5;
  const int wm   = wv / WGN;
  const int wn   = wv % WGN;
  const int half = lane >> 4;
  const int l16  = lane & 15;

  const int zb = blockIdx.z / nh;
  const int zh = blockIdx.z % nh;
  const long long aoff = (long long)zb * a_bstr + (long long)zh * a_hstr;
  const long long boff = (long long)zb * b_bstr + (long long)zh * b_hstr;
  const long long coff = (long long)zb * c_bstr + (long long)zh * c_hstr;

  const int rowBase = blockIdx.y * BM;
  const int colBase = blockIdx.x * BN;

  v8f acc[ITM][ITN];
#pragma unroll
  for (int tm = 0; tm < ITM; ++tm)
#pragma unroll
    for (int tn = 0; tn < ITN; ++tn) acc[tm][tn] = (v8f){};

  for (int k0 = 0; k0 < K; k0 += BK) {
    // ---- stage A tile (BM x BK) ----
#pragma unroll
    for (int ch = 0; ch < ACH; ++ch) {
      int idx = ch * (BDIM * 8) + tid * 8;
      int r = idx >> 6;          // / BK
      int c = idx & (BK - 1);
      if (A_F32) {
        const float* ap = (const float*)Ap + aoff + (long long)(rowBase + r) * lda + k0 + c;
        f32x4 v0 = *(const f32x4*)ap;
        f32x4 v1 = *(const f32x4*)(ap + 4);
        ushort4_t o0 = { f2bf(v0.x), f2bf(v0.y), f2bf(v0.z), f2bf(v0.w) };
        ushort4_t o1 = { f2bf(v1.x), f2bf(v1.y), f2bf(v1.z), f2bf(v1.w) };
        *(ushort4_t*)&sA[r * LSTR + c]     = o0;
        *(ushort4_t*)&sA[r * LSTR + c + 4] = o1;
        __builtin_prefetch(ap + BK, 0, 1);
      } else {
        const unsigned short* ap =
            (const unsigned short*)Ap + aoff + (long long)(rowBase + r) * lda + k0 + c;
#if USE_ASYNC_LDS
        async_cp16(ap, &sA[r * LSTR + c]);
#else
        *(u32x4*)&sA[r * LSTR + c] = *(const u32x4*)ap;
        __builtin_prefetch(ap + BK, 0, 1);
#endif
      }
    }
    // ---- stage B tile (BN x BK), source rows are [N][K]: contiguous 16B ----
#pragma unroll
    for (int ch = 0; ch < BCH; ++ch) {
      int idx = ch * (BDIM * 8) + tid * 8;
      int n = idx >> 6;
      int k = idx & (BK - 1);
      const unsigned short* bp = Bp + boff + (long long)(colBase + n) * ldb + k0 + k;
#if USE_ASYNC_LDS
      async_cp16(bp, &sB[n * LSTR + k]);
#else
      *(u32x4*)&sB[n * LSTR + k] = *(const u32x4*)bp;
      __builtin_prefetch(bp + BK, 0, 1);
#endif
    }
#if USE_ASYNC_LDS
    wait_async0();
#endif
    __syncthreads();

    // ---- WMMA over the two K=32 sub-steps ----
#pragma unroll
    for (int ks = 0; ks < BK; ks += 32) {
      union FB { v16bf v; unsigned int u[8]; };
      FB af[ITM], bfv[ITN];
#pragma unroll
      for (int tm = 0; tm < ITM; ++tm) {
        int row = wm * WTM + tm * 16 + l16;
#pragma unroll
        for (int v = 0; v < 8; ++v) {
          // ISA 16-bit A 16x32 layout: pairs (2v | 8+2v) + 8*half
          int kk = ks + half * 8 + ((v < 4) ? 2 * v : 8 + 2 * v);
          af[tm].u[v] = *(const unsigned int*)&sA[row * LSTR + kk];
        }
      }
#pragma unroll
      for (int tn = 0; tn < ITN; ++tn) {
        int n = wn * 64 + tn * 16 + l16;
#pragma unroll
        for (int v = 0; v < 8; ++v) {
          int kk = ks + half * 8 + ((v < 4) ? 2 * v : 8 + 2 * v);   // B 32x16: same K pairs as A
          bfv[tn].u[v] = *(const unsigned int*)&sB[n * LSTR + kk];
        }
      }
#pragma unroll
      for (int tm = 0; tm < ITM; ++tm)
#pragma unroll
        for (int tn = 0; tn < ITN; ++tn)
          acc[tm][tn] = __builtin_amdgcn_wmma_f32_16x16x32_bf16(
              false, af[tm].v, false, bfv[tn].v, (short)0, acc[tm][tn], false, false);
    }
    __syncthreads();
  }

  // ---- epilogue ----
#pragma unroll
  for (int tm = 0; tm < ITM; ++tm) {
#pragma unroll
    for (int tn = 0; tn < ITN; ++tn) {
      int col = colBase + wn * 64 + tn * 16 + l16;
#pragma unroll
      for (int r = 0; r < 8; ++r) {
        int row = rowBase + wm * WTM + tm * 16 + half * 8 + r;  // C/D VGPR layout
        float c = acc[tm][tn][r] * scale;
        if (EPI >= 1) c += bias[col];
        if (EPI == 2) c = 0.5f * c * (1.0f + erff(c * 0.70710678118654752f));  // exact GELU
        if (OUT == 2) {
          ((unsigned short*)Cp)[coff + (long long)col * ldc + row] = f2bf(c);
        } else {
          long long off = coff + (long long)row * ldc + col;
          if (OUT == 1) ((unsigned short*)Cp)[off] = f2bf(c);
          else          ((float*)Cp)[off] = c;
        }
      }
    }
  }
}

// --------------------- masked softmax over rows of S ------------------------
__global__ __launch_bounds__(BDIM) void softmax_mask(float* __restrict__ attn,
                                                     const int* __restrict__ mask) {
  const long long row = blockIdx.x;                 // [0, B*NH*S)
  const int b = (int)(row / ((long long)NHh * Ss));
  float* p = attn + row * (long long)Ss;
  const int* mrow = mask + (long long)b * Ss;
  __shared__ float red[BDIM];
  float v[8];
  float mx = -3.0e38f;
#pragma unroll
  for (int i = 0; i < 8; ++i) {
    int c = threadIdx.x + i * BDIM;
    float x = p[c];
    v[i] = (mrow[c] != 0) ? x : -3.0e38f;
    mx = fmaxf(mx, v[i]);
  }
  red[threadIdx.x] = mx; __syncthreads();
  for (int s = BDIM / 2; s > 0; s >>= 1) {
    if (threadIdx.x < s) red[threadIdx.x] = fmaxf(red[threadIdx.x], red[threadIdx.x + s]);
    __syncthreads();
  }
  mx = red[0]; __syncthreads();
  float sum = 0.f;
#pragma unroll
  for (int i = 0; i < 8; ++i) { v[i] = __expf(v[i] - mx); sum += v[i]; }
  red[threadIdx.x] = sum; __syncthreads();
  for (int s = BDIM / 2; s > 0; s >>= 1) {
    if (threadIdx.x < s) red[threadIdx.x] += red[threadIdx.x + s];
    __syncthreads();
  }
  float inv = 1.0f / red[0];
#pragma unroll
  for (int i = 0; i < 8; ++i) p[threadIdx.x + i * BDIM] = v[i] * inv;
}

// --------------------- residual add + LayerNorm -----------------------------
template <int EMIT_BF>
__global__ __launch_bounds__(BDIM) void add_layernorm(
    const float* __restrict__ X, const float* __restrict__ Y,
    const float* __restrict__ g, const float* __restrict__ be,
    float* __restrict__ outF, unsigned short* __restrict__ outB) {
  const long long base = (long long)blockIdx.x * Hh;
  __shared__ float r1[BDIM], r2[BDIM];
  float v[4]; float s1 = 0.f, s2 = 0.f;
#pragma unroll
  for (int i = 0; i < 4; ++i) {
    int c = threadIdx.x + i * BDIM;
    float x = X[base + c] + Y[base + c];
    v[i] = x; s1 += x; s2 += x * x;
  }
  r1[threadIdx.x] = s1; r2[threadIdx.x] = s2; __syncthreads();
  for (int s = BDIM / 2; s > 0; s >>= 1) {
    if (threadIdx.x < s) { r1[threadIdx.x] += r1[threadIdx.x + s]; r2[threadIdx.x] += r2[threadIdx.x + s]; }
    __syncthreads();
  }
  float mu  = r1[0] * (1.0f / Hh);
  float var = r2[0] * (1.0f / Hh) - mu * mu;
  float rstd = rsqrtf(var + 1e-5f);
#pragma unroll
  for (int i = 0; i < 4; ++i) {
    int c = threadIdx.x + i * BDIM;
    float o = (v[i] - mu) * rstd * g[c] + be[c];
    outF[base + c] = o;
    if (EMIT_BF) outB[base + c] = f2bf(o);
  }
}

// ---------------------------------------------------------------------------
extern "C" void kernel_launch(void* const* d_in, const int* in_sizes, int n_in,
                              void* d_out, int out_size, void* d_ws, size_t ws_size,
                              hipStream_t stream) {
  const float* x   = (const float*)d_in[0];
  const int*   msk = (const int*)d_in[1];
  const float* Wq  = (const float*)d_in[2];
  const float* bq  = (const float*)d_in[3];
  const float* Wk  = (const float*)d_in[4];
  const float* bk  = (const float*)d_in[5];
  const float* Wv  = (const float*)d_in[6];
  const float* bv  = (const float*)d_in[7];
  const float* Wo  = (const float*)d_in[8];
  const float* bo  = (const float*)d_in[9];
  const float* W1  = (const float*)d_in[10];
  const float* b1  = (const float*)d_in[11];
  const float* W2  = (const float*)d_in[12];
  const float* b2  = (const float*)d_in[13];
  const float* g1  = (const float*)d_in[14];
  const float* be1 = (const float*)d_in[15];
  const float* g2  = (const float*)d_in[16];
  const float* be2 = (const float*)d_in[17];

  const long long E = (long long)Bb * Ss * Hh;          // 4,194,304
  const int M = Bb * Ss;                                // 4096
  float* outX = (float*)d_out;
  float* attn = outX + E;                               // [B,NH,S,S] fp32

  char* ws = (char*)d_ws;
  size_t off = 0;
  auto alloc = [&](size_t bytes) { void* p = ws + off; off += (bytes + 255) & ~(size_t)255; return p; };
  unsigned short* xb   = (unsigned short*)alloc(E * 2);
  unsigned short* WqT  = (unsigned short*)alloc((size_t)Hh * Hh * 2);   // [N][K]
  unsigned short* WkT  = (unsigned short*)alloc((size_t)Hh * Hh * 2);
  unsigned short* WvT  = (unsigned short*)alloc((size_t)Hh * Hh * 2);
  unsigned short* WoT  = (unsigned short*)alloc((size_t)Hh * Hh * 2);
  unsigned short* W1T  = (unsigned short*)alloc((size_t)Hh * FFf * 2);  // [FF][H]
  unsigned short* W2T  = (unsigned short*)alloc((size_t)FFf * Hh * 2);  // [H][FF]
  unsigned short* Qb   = (unsigned short*)alloc(E * 2);                 // [B*S][H]
  unsigned short* Kb   = (unsigned short*)alloc(E * 2);                 // [B*S][H]
  unsigned short* vt   = (unsigned short*)alloc(E * 2);                 // V^T [H][B*S]
  unsigned short* ctxb = (unsigned short*)alloc(E * 2);                 // [B*S][H]
  unsigned short* x1b  = (unsigned short*)alloc(E * 2);
  unsigned short* hb   = (unsigned short*)alloc((size_t)Bb * Ss * FFf * 2);
  float* t0 = (float*)alloc(E * 4);   // attn_out
  float* t1 = (float*)alloc(E * 4);   // x1 fp32
  float* t2 = (float*)alloc(E * 4);   // ffn fp32

  auto cast = [&](const float* src, unsigned short* dst, long long n) {
    int n4 = (int)(n / 4);
    cast_f32_bf16<<<(n4 + BDIM - 1) / BDIM, BDIM, 0, stream>>>(src, dst, n4);
  };
  auto castT = [&](const float* src, unsigned short* dst, int R, int C) {
    cast_transpose<<<dim3(C / 32, R / 32), BDIM, 0, stream>>>(src, dst, R, C);
  };
  cast(x, xb, E);
  castT(Wq, WqT, Hh, Hh);
  castT(Wk, WkT, Hh, Hh);
  castT(Wv, WvT, Hh, Hh);
  castT(Wo, WoT, Hh, Hh);
  castT(W1, W1T, Hh, FFf);
  castT(W2, W2T, FFf, Hh);

  const long long SH = (long long)Ss * Hh, SS = (long long)Ss * Ss;

  // Q/K = x @ W + b (bf16, [B*S][H]);  V written transposed (V^T [H][B*S])
  dim3 gQKV(Hh / 128, M / 128, 1);
  gemm_wmma<128, 1, 1, 0><<<gQKV, BDIM, 0, stream>>>(
      xb, WqT, bq, Qb, Hh, Hh, Hh, Hh, 0, 0, 0, 0, 0, 0, 1, 1.0f);
  gemm_wmma<128, 1, 1, 0><<<gQKV, BDIM, 0, stream>>>(
      xb, WkT, bk, Kb, Hh, Hh, Hh, Hh, 0, 0, 0, 0, 0, 0, 1, 1.0f);
  gemm_wmma<128, 1, 2, 0><<<gQKV, BDIM, 0, stream>>>(
      xb, WvT, bv, vt, Hh, Hh, Hh, M /*ldc of V^T*/, 0, 0, 0, 0, 0, 0, 1, 1.0f);

  // scores = Q @ K^T / 8 -> attn region (fp32, raw); B rows = K[b,t,h,:]
  gemm_wmma<128, 0, 0, 0><<<dim3(Ss / 128, Ss / 128, Bb * NHh), BDIM, 0, stream>>>(
      Qb, Kb, nullptr, attn, DHh, Hh, Hh, Ss,
      SH, 64, SH, 64, (long long)NHh * SS, SS, NHh, 0.125f);

  // softmax with mask, in place
  softmax_mask<<<(unsigned)(Bb * NHh * Ss), BDIM, 0, stream>>>(attn, msk);

  // ctx = attn @ V : B rows = V^T[h*64+d][b*S + t], ldb = B*S
  gemm_wmma<64, 0, 1, 1><<<dim3(1, Ss / 128, Bb * NHh), BDIM, 0, stream>>>(
      attn, vt, nullptr, ctxb, Ss, Ss, M, Hh,
      (long long)NHh * SS, SS, (long long)Ss, (long long)DHh * M, SH, 64, NHh, 1.0f);

  // attn_out = ctx @ Wo + bo (fp32)
  gemm_wmma<128, 1, 0, 0><<<dim3(Hh / 128, M / 128, 1), BDIM, 0, stream>>>(
      ctxb, WoT, bo, t0, Hh, Hh, Hh, Hh, 0, 0, 0, 0, 0, 0, 1, 1.0f);

  // x1 = LN(x + attn_out)  (fp32 + bf16)
  add_layernorm<1><<<M, BDIM, 0, stream>>>(x, t0, g1, be1, t1, x1b);

  // h = GELU(x1 @ W1 + b1) (bf16)
  gemm_wmma<128, 2, 1, 0><<<dim3(FFf / 128, M / 128, 1), BDIM, 0, stream>>>(
      x1b, W1T, b1, hb, Hh, Hh, Hh, FFf, 0, 0, 0, 0, 0, 0, 1, 1.0f);

  // ffn = h @ W2 + b2 (fp32)
  gemm_wmma<128, 1, 0, 0><<<dim3(Hh / 128, M / 128, 1), BDIM, 0, stream>>>(
      hb, W2T, b2, t2, FFf, FFf, FFf, Hh, 0, 0, 0, 0, 0, 0, 1, 1.0f);

  // x2 = LN(x1 + ffn) -> d_out
  add_layernorm<0><<<M, BDIM, 0, stream>>>(t1, t2, g2, be2, outX, nullptr);
}
